// LayerBlock_33105607917791
// MI455X (gfx1250) — matrix operand order repaired and checked by reading, not compiled, not run on hardware
//
#include <hip/hip_runtime.h>
#include <cstdint>

// ---------------------------------------------------------------------------
// Structural constants of the reference problem (LMAX=6, MMAX=2, C=128, H=256)
// ---------------------------------------------------------------------------
#define NRED       29          // reduced coefficients (|m|<=2)
#define NFULL      49          // (LMAX+1)^2
#define CCH        128         // channels C
#define HH         256         // hidden H
#define ECH        128         // edge channels
#define NGAUSS     512
#define RED_STRIDE (NRED * CCH)    // 3712
#define FULL_STRIDE (NFULL * CCH)  // 6272
#define WIG_STRIDE (NFULL * NFULL) // 2401

typedef __bf16 bf16;
typedef __bf16 v16bf __attribute__((ext_vector_type(16)));
typedef float  v8f   __attribute__((ext_vector_type(8)));

// PERM groups reduced coefficients by m: [m=0 (7)],[m=+1 (6)],[m=-1 (6)],[m=+2 (5)],[m=-2 (5)]
__constant__ int c_PERM[NRED] = {0,2,6,11,16,21,26, 3,7,12,17,22,27, 1,5,10,15,20,25,
                                 8,13,18,23,28, 4,9,14,19,24};
// OUT_MASK composed with PERM: full-index rows of wigner in PERM order
__constant__ int c_OMP[NRED]  = {0,2,6,12,20,30,42, 3,7,13,21,31,43, 1,5,11,19,29,41,
                                 8,14,22,32,44, 4,10,18,28,40};
// OUT_MASK in original reduced order
__constant__ int c_OM[NRED]   = {0,1,2,3,4,5,6,7,8, 10,11,12,13,14, 18,19,20,21,22,
                                 28,29,30,31,32, 40,41,42,43,44};

__device__ __forceinline__ float siluf(float x) { return x / (1.0f + __expf(-x)); }

// ---------------------------------------------------------------------------
// CDNA5 async global->LDS copy (byte-exact staging, tracked by ASYNCcnt).
// Each lane copies 16 contiguous bytes: LDS[lds_byte_off] = GLOBAL[gptr].
// ---------------------------------------------------------------------------
__device__ __forceinline__ void async_g2l_b128(uint32_t lds_byte_off, const void* gptr) {
  asm volatile("global_load_async_to_lds_b128 %0, %1, off"
               :: "v"(lds_byte_off), "v"(gptr)
               : "memory");
}
__device__ __forceinline__ void wait_asynccnt0() {
  asm volatile("s_wait_asynccnt 0x0" ::: "memory");
}

// ---------------------------------------------------------------------------
// WMMA helpers (gfx1250, wave32, 16x16x32 bf16, f32 accumulate)
// A fragment: LDS row-major [16][ld] bf16 tile.
// B fragment: LDS "Bt" layout [16 n-rows][ld] bf16 (i.e. B transposed: n major, k minor)
// ---------------------------------------------------------------------------
__device__ __forceinline__ v16bf lda16(const bf16* a, int ld) {
  int lane = threadIdx.x & 31;
  int r = lane & 15, hi = lane >> 4;
  const bf16* p = a + r * ld;
  v16bf v;
#pragma unroll
  for (int j = 0; j < 4; ++j) {
    int k0 = hi * 8 + 2 * j;
    v[2*j]     = p[k0];
    v[2*j + 1] = p[k0 + 1];
    int k1 = 16 + hi * 8 + 2 * j;
    v[8 + 2*j]     = p[k1];
    v[8 + 2*j + 1] = p[k1 + 1];
  }
  return v;
}

__device__ __forceinline__ v16bf ldb16(const bf16* bt, int ld) {
  int lane = threadIdx.x & 31;
  int n = lane & 15, hi = lane >> 4;
  const bf16* p = bt + n * ld + hi * 8;
  v16bf v;
#pragma unroll
  for (int j = 0; j < 8; ++j) { int k = (j < 4) ? 2*j : 16 + 2*(j - 4); v[2*j] = p[k]; v[2*j + 1] = p[k + 1]; }
  return v;
}

__device__ __forceinline__ v8f wmma_bf(v16bf a, v16bf b, v8f c) {
  return __builtin_amdgcn_wmma_f32_16x16x32_bf16(false, a, false, b, (short)0, c, false, false);
}

// ---------------------------------------------------------------------------
// K1: edge embedding  x_edge[E][128]
// ---------------------------------------------------------------------------
__global__ void edge_embed(const float* __restrict__ dist, const int* __restrict__ eidx,
                           const int* __restrict__ zat,
                           const float* __restrict__ w1, const float* __restrict__ b1,
                           const float* __restrict__ semb, const float* __restrict__ temb,
                           const float* __restrict__ w2, const float* __restrict__ b2,
                           float* __restrict__ xe, int NE) {
  __shared__ float gs[NGAUSS];
  __shared__ float t1[ECH];
  int e = blockIdx.x, tid = threadIdx.x;
  float d = dist[e];
  const float step  = 8.0f / 511.0f;
  const float coeff = -0.5f / (step * step);
  for (int g = tid; g < NGAUSS; g += 128) {
    float o = (float)g * step;
    gs[g] = __expf(coeff * (d - o) * (d - o));
  }
  __syncthreads();
  int src = eidx[e], tgt = eidx[NE + e];
  float acc = b1[tid];
  const float* wr = w1 + tid * NGAUSS;
  for (int k = 0; k < NGAUSS; ++k) acc += gs[k] * wr[k];
  acc += semb[zat[src] * ECH + tid] + temb[zat[tgt] * ECH + tid];
  t1[tid] = siluf(acc);
  __syncthreads();
  float acc2 = b2[tid];
  const float* wr2 = w2 + tid * ECH;
  for (int k = 0; k < ECH; ++k) acc2 += t1[k] * wr2[k];
  xe[e * ECH + tid] = siluf(acc2);
}

// ---------------------------------------------------------------------------
// K2: per-edge Wigner rotation (PERM order output, bf16):
//   xm_s[e] = wig[e][OUT_MASK[PERM]] (29x49) @ x[node] (49x128)
// ---------------------------------------------------------------------------
__global__ void rotate_fwd(const float* __restrict__ x, const float* __restrict__ wig,
                           const int* __restrict__ eidx, int NE,
                           bf16* __restrict__ xm_src, bf16* __restrict__ xm_tgt) {
  __shared__ bf16 Awm[32 * 64];
  __shared__ bf16 Bts[128 * 64];
  __shared__ bf16 Btt[128 * 64];
  int e = blockIdx.x, tid = threadIdx.x;
  int src = eidx[e], tgt = eidx[NE + e];
  for (int t = tid; t < 32 * 64; t += 256) {
    int j = t >> 6, k = t & 63;
    Awm[t] = (j < NRED && k < NFULL) ? (bf16)wig[(size_t)e * WIG_STRIDE + c_OMP[j] * NFULL + k]
                                     : (bf16)0.0f;
  }
  for (int t = tid; t < 128 * 64; t += 256) {
    int n = t >> 6, k = t & 63;
    Bts[t] = (k < NFULL) ? (bf16)x[(size_t)src * FULL_STRIDE + k * CCH + n] : (bf16)0.0f;
    Btt[t] = (k < NFULL) ? (bf16)x[(size_t)tgt * FULL_STRIDE + k * CCH + n] : (bf16)0.0f;
  }
  __syncthreads();
  int w = tid >> 5, lane = tid & 31, nl = lane & 15, hi = lane >> 4;
  for (int s = 0; s < 2; ++s) {
    const bf16* Bt = s ? Btt : Bts;
    bf16* out = (s ? xm_tgt : xm_src) + (size_t)e * RED_STRIDE;
    for (int mt = 0; mt < 2; ++mt) {
      v8f acc = {};
#pragma unroll
      for (int kc = 0; kc < 2; ++kc) {
        v16bf a = lda16(Awm + mt * 16 * 64 + kc * 32, 64);
        v16bf b = ldb16(Bt + w * 16 * 64 + kc * 32, 64);
        acc = wmma_bf(a, b, acc);
      }
#pragma unroll
      for (int i = 0; i < 8; ++i) {
        int row = mt * 16 + hi * 8 + i;
        if (row < NRED) out[row * CCH + w * 16 + nl] = (bf16)acc[i];
      }
    }
  }
}

// ---------------------------------------------------------------------------
// SO2 stage-1 GEMM: T[rows][256] = A[rows][K] @ W[256][K]^T
//  A rows are a strided view into xm (rpe components per edge).
//  A tiles are bf16 byte-copies -> staged with GLOBAL_LOAD_ASYNC_TO_LDS_B128.
// ---------------------------------------------------------------------------
__global__ void so2_gemm1(const bf16* __restrict__ A, int rows, int rpe, int off, int cstride,
                          int K, const float* __restrict__ W, float* __restrict__ T) {
  __shared__ __align__(16) bf16 As[64 * 32];
  __shared__ bf16 Bs[128 * 32];
  int tid = threadIdx.x, w = tid >> 5, lane = tid & 31, nl = lane & 15, hi = lane >> 4;
  int bm = blockIdx.x * 64, bn = blockIdx.y * 128;
  v8f acc[4] = {{}, {}, {}, {}};
  int nk = K >> 5;
  bool fullA = (bm + 64 <= rows);
  uint32_t as_lds = (uint32_t)(uintptr_t)(&As[0]) + (uint32_t)tid * 16u;
  // this thread's fixed A row/col for the async path (8 bf16 = 16B per lane)
  int arl = tid >> 2, akk = (tid & 3) * 8;
  int aR = bm + arl;
  const bf16* arow = A + (size_t)(aR / rpe) * RED_STRIDE + off + (aR % rpe) * cstride + akk;
  for (int kc = 0; kc < nk; ++kc) {
    if (fullA) {
      async_g2l_b128(as_lds, arow + kc * 32);
    } else {
      for (int t = tid; t < 64 * 32; t += 256) {
        int rl = t >> 5, kk = t & 31;
        int R = bm + rl;
        As[t] = (R < rows)
                    ? A[(size_t)(R / rpe) * RED_STRIDE + off + (R % rpe) * cstride + kc * 32 + kk]
                    : (bf16)0.0f;
      }
    }
    if (kc + 1 < nk)  // prefetch next weight chunk (L2-resident stream)
      __builtin_prefetch(&W[(size_t)(bn + (tid >> 1)) * K + (kc + 1) * 32], 0, 3);
    for (int t = tid; t < 128 * 32; t += 256) {
      int j = t >> 5, kk = t & 31;
      Bs[t] = (bf16)W[(size_t)(bn + j) * K + kc * 32 + kk];
    }
    wait_asynccnt0();
    __syncthreads();
    v16bf b = ldb16(Bs + w * 16 * 32, 32);
#pragma unroll
    for (int mt = 0; mt < 4; ++mt) {
      v16bf a = lda16(As + mt * 16 * 32, 32);
      acc[mt] = wmma_bf(a, b, acc[mt]);
    }
    __syncthreads();
  }
  int gcol = bn + w * 16 + nl;
#pragma unroll
  for (int mt = 0; mt < 4; ++mt)
#pragma unroll
    for (int i = 0; i < 8; ++i) {
      int R = bm + mt * 16 + hi * 8 + i;
      if (R < rows) T[(size_t)R * HH + gcol] = acc[mt][i];
    }
}

// ---------------------------------------------------------------------------
// SO2 gate: T[r][h] *= silu(x_edge[e] . gw[gateOff+h] + gb[gateOff+h]),  e=r/rpe
// ---------------------------------------------------------------------------
__global__ void so2_gate(float* __restrict__ T, const float* __restrict__ xe,
                         const float* __restrict__ gw, const float* __restrict__ gb,
                         int gateOff, int rpe) {
  __shared__ float xs[ECH];
  int e = blockIdx.x, tid = threadIdx.x;
  if (tid < ECH) xs[tid] = xe[e * ECH + tid];
  __syncthreads();
  float acc = gb[gateOff + tid];
  const float* wr = gw + (size_t)(gateOff + tid) * ECH;
  for (int k = 0; k < ECH; ++k) acc += xs[k] * wr[k];
  float g = siluf(acc);
  for (int c = 0; c < rpe; ++c) T[((size_t)e * rpe + c) * HH + tid] *= g;
}

// ---------------------------------------------------------------------------
// SO2 stage-2 GEMM: msg (+)= remap( T[rows][256] @ W2[N][256]^T )
//  flip=0: dest comp = comp, sign +.  flip=1 (imag path): dest comp = 1-comp,
//  sign = -1 for comp==1 (implements out0 = xr0 - xi1 ; out1 = xr1 + xi0)
// ---------------------------------------------------------------------------
__global__ void so2_gemm2(const float* __restrict__ T, int rows, int N,
                          const float* __restrict__ W2, float* __restrict__ msg,
                          int rpe, int off, int cstride, int flip) {
  __shared__ bf16 As[64 * 32];
  __shared__ bf16 Bs[128 * 32];
  int tid = threadIdx.x, w = tid >> 5, lane = tid & 31, nl = lane & 15, hi = lane >> 4;
  int bm = blockIdx.x * 64, bn = blockIdx.y * 128;
  v8f acc[4] = {{}, {}, {}, {}};
  for (int kc = 0; kc < HH / 32; ++kc) {
    for (int t = tid; t < 64 * 32; t += 256) {
      int rl = t >> 5, kk = t & 31;
      int R = bm + rl;
      As[t] = (R < rows) ? (bf16)T[(size_t)R * HH + kc * 32 + kk] : (bf16)0.0f;
    }
    if (kc + 1 < HH / 32)
      __builtin_prefetch(&W2[(size_t)(bn + (tid >> 1)) * HH + (kc + 1) * 32], 0, 3);
    for (int t = tid; t < 128 * 32; t += 256) {
      int j = t >> 5, kk = t & 31;
      Bs[t] = (bf16)W2[(size_t)(bn + j) * HH + kc * 32 + kk];
    }
    __syncthreads();
    v16bf b = ldb16(Bs + w * 16 * 32, 32);
#pragma unroll
    for (int mt = 0; mt < 4; ++mt) {
      v16bf a = lda16(As + mt * 16 * 32, 32);
      acc[mt] = wmma_bf(a, b, acc[mt]);
    }
    __syncthreads();
  }
  int gcol = bn + w * 16 + nl;
#pragma unroll
  for (int mt = 0; mt < 4; ++mt)
#pragma unroll
    for (int i = 0; i < 8; ++i) {
      int R = bm + mt * 16 + hi * 8 + i;
      if (R < rows) {
        int e = R / rpe, c = R % rpe;
        int dc = flip ? 1 - c : c;
        float sg = (flip && c == 1) ? -1.0f : 1.0f;
        msg[(size_t)e * RED_STRIDE + off + dc * cstride + gcol] += sg * acc[mt][i];
      }
    }
}

// ---------------------------------------------------------------------------
// K5: per-edge m-grid nonlinearity, fused:
//   G = silu( Tg(perm cols)[70x29] @ msg_e[29x128] ) ;  msg2_e = Fg^T[29x70] @ G
// msg in PERM order; msg2 written in ORIGINAL reduced order (bf16).
// ---------------------------------------------------------------------------
__global__ void grid_mm(const float* __restrict__ msg, const float* __restrict__ tg,
                        const float* __restrict__ fg, bf16* __restrict__ msg2) {
  __shared__ bf16 Tg[80 * 32];
  __shared__ bf16 Btm[128 * 32];
  __shared__ bf16 Bt2[128 * 96];
  __shared__ bf16 A2[32 * 96];
  int e = blockIdx.x, tid = threadIdx.x;
  int w = tid >> 5, lane = tid & 31, nl = lane & 15, hi = lane >> 4;
  for (int t = tid; t < 80 * 32; t += 256) {
    int r = t >> 5, k = t & 31;
    Tg[t] = (r < 70 && k < NRED) ? (bf16)tg[r * NRED + c_PERM[k]] : (bf16)0.0f;
  }
  for (int t = tid; t < 128 * 32; t += 256) {
    int n = t >> 5, k = t & 31;
    Btm[t] = (k < NRED) ? (bf16)msg[(size_t)e * RED_STRIDE + k * CCH + n] : (bf16)0.0f;
  }
  for (int t = tid; t < 128 * 96; t += 256) Bt2[t] = (bf16)0.0f;
  for (int t = tid; t < 32 * 96; t += 256) {
    int i = t / 96, k = t % 96;
    A2[t] = (i < NRED && k < 70) ? (bf16)fg[k * NRED + i] : (bf16)0.0f;
  }
  __syncthreads();
  // GEMM1 + silu -> Bt2 (transposed, bf16)
  v16bf b1 = ldb16(Btm + w * 16 * 32, 32);
  for (int mt = 0; mt < 5; ++mt) {
    v8f acc = {};
    v16bf a = lda16(Tg + mt * 16 * 32, 32);
    acc = wmma_bf(a, b1, acc);
#pragma unroll
    for (int i = 0; i < 8; ++i) {
      int grow = mt * 16 + hi * 8 + i;
      if (grow < 70) Bt2[(w * 16 + nl) * 96 + grow] = (bf16)siluf(acc[i]);
    }
  }
  __syncthreads();
  // GEMM2 -> msg2 original order
  for (int mt = 0; mt < 2; ++mt) {
    v8f acc = {};
#pragma unroll
    for (int kc = 0; kc < 3; ++kc) {
      v16bf a = lda16(A2 + mt * 16 * 96 + kc * 32, 96);
      v16bf b = ldb16(Bt2 + w * 16 * 96 + kc * 32, 96);
      acc = wmma_bf(a, b, acc);
    }
#pragma unroll
    for (int i = 0; i < 8; ++i) {
      int grow = mt * 16 + hi * 8 + i;
      if (grow < NRED) msg2[(size_t)e * RED_STRIDE + grow * CCH + w * 16 + nl] = (bf16)acc[i];
    }
  }
}

// ---------------------------------------------------------------------------
// K6: inverse rotation + segment sum:
//   nmsg[tgt] += wig^T[:,OUT_MASK] (49x29) @ msg2_e (29x128)
// ---------------------------------------------------------------------------
__global__ void winv_scatter(const bf16* __restrict__ msg2, const float* __restrict__ wig,
                             const int* __restrict__ eidx, int NE, float* __restrict__ nmsg) {
  __shared__ bf16 Ai[64 * 32];
  __shared__ bf16 Bt[128 * 32];
  int e = blockIdx.x, tid = threadIdx.x;
  int w = tid >> 5, lane = tid & 31, nl = lane & 15, hi = lane >> 4;
  int tgt = eidx[NE + e];
  for (int t = tid; t < 64 * 32; t += 256) {
    int i = t >> 5, j = t & 31;
    Ai[t] = (i < NFULL && j < NRED) ? (bf16)wig[(size_t)e * WIG_STRIDE + c_OM[j] * NFULL + i]
                                    : (bf16)0.0f;
  }
  for (int t = tid; t < 128 * 32; t += 256) {
    int n = t >> 5, j = t & 31;
    Bt[t] = (j < NRED) ? msg2[(size_t)e * RED_STRIDE + j * CCH + n] : (bf16)0.0f;
  }
  __syncthreads();
  v16bf b = ldb16(Bt + w * 16 * 32, 32);
  for (int mt = 0; mt < 4; ++mt) {
    v8f acc = {};
    v16bf a = lda16(Ai + mt * 16 * 32, 32);
    acc = wmma_bf(a, b, acc);
#pragma unroll
    for (int i = 0; i < 8; ++i) {
      int row = mt * 16 + hi * 8 + i;
      if (row < NFULL)
        atomicAdd(&nmsg[(size_t)tgt * FULL_STRIDE + row * CCH + w * 16 + nl], acc[i]);
    }
  }
}

// ---------------------------------------------------------------------------
// K7: per-node ll-grid + sphere MLP + from-grid, fully fused in LDS.
// ---------------------------------------------------------------------------
__global__ void node_sphere(const float* __restrict__ x, const float* __restrict__ nmsg,
                            const float* __restrict__ tll, const float* __restrict__ fll,
                            const float* __restrict__ fw1, const float* __restrict__ fb1,
                            const float* __restrict__ fw2, const float* __restrict__ fb2,
                            const float* __restrict__ fw3, const float* __restrict__ fb3,
                            float* __restrict__ out) {
  __shared__ bf16 Btx[128 * 64];   // x[node] transposed, k-padded
  __shared__ bf16 Btm[128 * 64];   // nmsg[node] transposed
  __shared__ bf16 GcH2[16 * 256];  // grid chunk [16][256] (gx|gm); later H2 [16][128]
  __shared__ bf16 BwA3[128 * 32];  // per-chunk weight B tiles; later A3 [64][32]
  __shared__ bf16 A2g[2 * 16 * 32];
  __shared__ bf16 H1[16 * 128];
  __shared__ bf16 H3t[128 * 32];   // fc3 out transposed (cols 16..31 stay zero)
  int node = blockIdx.x, tid = threadIdx.x;
  int w = tid >> 5, lane = tid & 31, nl = lane & 15, hi = lane >> 4;
  for (int t = tid; t < 128 * 64; t += 256) {
    int n = t >> 6, k = t & 63;
    Btx[t] = (k < NFULL) ? (bf16)x[(size_t)node * FULL_STRIDE + k * CCH + n] : (bf16)0.0f;
    Btm[t] = (k < NFULL) ? (bf16)nmsg[(size_t)node * FULL_STRIDE + k * CCH + n] : (bf16)0.0f;
  }
  for (int t = tid; t < 128 * 32; t += 256) H3t[t] = (bf16)0.0f;
  v8f outAcc[4] = {{}, {}, {}, {}};
  int col = w * 16 + nl;
  for (int ch = 0; ch < 13; ++ch) {
    int m0 = ch * 16;
    // --- stage to_grid_ll chunk rows
    for (int t = tid; t < 2 * 16 * 32; t += 256) {
      int kc = t >> 9, r = (t >> 5) & 15, k = t & 31;
      int grow = m0 + r, gk = kc * 32 + k;
      A2g[t] = (grow < 196 && gk < NFULL) ? (bf16)tll[grow * NFULL + gk] : (bf16)0.0f;
    }
    __syncthreads();
    // --- gx / gm projection -> Gc [16][256]
    for (int sel = 0; sel < 2; ++sel) {
      const bf16* Bt = sel ? Btm : Btx;
      v8f acc = {};
#pragma unroll
      for (int kc = 0; kc < 2; ++kc) {
        v16bf a = lda16(A2g + kc * 16 * 32, 32);
        v16bf b = ldb16(Bt + w * 16 * 64 + kc * 32, 64);
        acc = wmma_bf(a, b, acc);
      }
#pragma unroll
      for (int i = 0; i < 8; ++i) {
        int row = hi * 8 + i;
        GcH2[row * 256 + sel * 128 + col] = (bf16)acc[i];
      }
    }
    __syncthreads();
    // --- fc1 (K=256) -> H1, silu
    {
      v8f acc = {};
      for (int kc = 0; kc < 8; ++kc) {
        for (int t = tid; t < 128 * 32; t += 256) {
          int j = t >> 5, kk = t & 31;
          BwA3[t] = (bf16)fw1[(size_t)j * 256 + kc * 32 + kk];
        }
        __syncthreads();
        v16bf a = lda16(GcH2 + kc * 32, 256);
        v16bf b = ldb16(BwA3 + w * 16 * 32, 32);
        acc = wmma_bf(a, b, acc);
        __syncthreads();
      }
#pragma unroll
      for (int i = 0; i < 8; ++i) {
        int row = hi * 8 + i;
        H1[row * 128 + col] = (bf16)siluf(acc[i] + fb1[col]);
      }
    }
    __syncthreads();
    // --- fc2 (K=128) -> H2 (into GcH2), silu
    {
      v8f acc = {};
      for (int kc = 0; kc < 4; ++kc) {
        for (int t = tid; t < 128 * 32; t += 256) {
          int j = t >> 5, kk = t & 31;
          BwA3[t] = (bf16)fw2[(size_t)j * 128 + kc * 32 + kk];
        }
        __syncthreads();
        v16bf a = lda16(H1 + kc * 32, 128);
        v16bf b = ldb16(BwA3 + w * 16 * 32, 32);
        acc = wmma_bf(a, b, acc);
        __syncthreads();
      }
#pragma unroll
      for (int i = 0; i < 8; ++i) {
        int row = hi * 8 + i;
        GcH2[row * 128 + col] = (bf16)siluf(acc[i] + fb2[col]);
      }
    }
    __syncthreads();
    // --- fc3 (K=128) -> H3t transposed, + bias (no silu)
    {
      v8f acc = {};
      for (int kc = 0; kc < 4; ++kc) {
        for (int t = tid; t < 128 * 32; t += 256) {
          int j = t >> 5, kk = t & 31;
          BwA3[t] = (bf16)fw3[(size_t)j * 128 + kc * 32 + kk];
        }
        __syncthreads();
        v16bf a = lda16(GcH2 + kc * 32, 128);
        v16bf b = ldb16(BwA3 + w * 16 * 32, 32);
        acc = wmma_bf(a, b, acc);
        __syncthreads();
      }
#pragma unroll
      for (int i = 0; i < 8; ++i) {
        int row = hi * 8 + i;
        H3t[col * 32 + row] = (bf16)(acc[i] + fb3[col]);
      }
    }
    __syncthreads();
    // --- accumulate out49 += from_ll_chunk^T @ H3
    for (int t = tid; t < 64 * 32; t += 256) {
      int i2 = t >> 5, k = t & 31;
      int grow = m0 + k;
      BwA3[t] = (k < 16 && grow < 196 && i2 < NFULL) ? (bf16)fll[grow * NFULL + i2]
                                                     : (bf16)0.0f;
    }
    __syncthreads();
    v16bf b = ldb16(H3t + w * 16 * 32, 32);
#pragma unroll
    for (int mt = 0; mt < 4; ++mt) {
      v16bf a = lda16(BwA3 + mt * 16 * 32, 32);
      outAcc[mt] = wmma_bf(a, b, outAcc[mt]);
    }
    __syncthreads();
  }
#pragma unroll
  for (int mt = 0; mt < 4; ++mt)
#pragma unroll
    for (int i = 0; i < 8; ++i) {
      int row = mt * 16 + hi * 8 + i;
      if (row < NFULL) out[(size_t)node * FULL_STRIDE + row * CCH + col] = outAcc[mt][i];
    }
}

// ---------------------------------------------------------------------------
// Host launcher
// Flat input order (top-level dict insertion order; nested dicts/lists in
// insertion order):
//  0 x, 1 atomic_numbers, 2 edge_distance, 3 edge_index, 4 wigner,
//  5..10 edge_params {fc1_dist_w,fc1_dist_b,src_emb,tgt_emb,fc1_edge_w,fc1_edge_b},
//  11..26 so2_src {fc1_dist0_w,fc1_dist0_b,fc1_m0,fc2_m0,
//                  m1:{fc1_dist_w,fc1_dist_b,fc1_r,fc2_r,fc1_i,fc2_i},
//                  m2:{...}},
//  27..42 so2_tgt (same), 43..48 sphere {fc1_w,fc1_b,fc2_w,fc2_b,fc3_w,fc3_b},
//  49 to_grid_mm, 50 from_grid_mm, 51 to_grid_ll, 52 from_grid_ll
// ---------------------------------------------------------------------------
static inline int cdiv(int a, int b) { return (a + b - 1) / b; }

extern "C" void kernel_launch(void* const* d_in, const int* in_sizes, int n_in,
                              void* d_out, int out_size, void* d_ws, size_t ws_size,
                              hipStream_t stream) {
  const int NE = in_sizes[2];  // edges
  const int NN = in_sizes[1];  // nodes
  const float* x    = (const float*)d_in[0];
  const int*   zat  = (const int*)d_in[1];
  const float* dist = (const float*)d_in[2];
  const int*   eidx = (const int*)d_in[3];
  const float* wig  = (const float*)d_in[4];
  const float* tgmm = (const float*)d_in[49];
  const float* fgmm = (const float*)d_in[50];
  const float* tgll = (const float*)d_in[51];
  const float* fgll = (const float*)d_in[52];

  // workspace carve (~437 MB)
  char* base = (char*)d_ws;
  size_t off = 0;
  auto carve = [&](size_t bytes) -> char* {
    char* r = base + off;
    off = (off + bytes + 255) & ~(size_t)255;
    return r;
  };
  float* xe   = (float*)carve((size_t)NE * ECH * 4);
  float* T    = (float*)carve((size_t)2 * NE * HH * 4);
  float* msg  = (float*)carve((size_t)NE * RED_STRIDE * 4);
  float* nmsg = (float*)carve((size_t)NN * FULL_STRIDE * 4);
  bf16* xmsrc = (bf16*)carve((size_t)NE * RED_STRIDE * 2);
  bf16* xmtgt = (bf16*)carve((size_t)NE * RED_STRIDE * 2);
  bf16* msg2  = xmsrc;  // alias: xm_src dead before grid_mm runs

  edge_embed<<<NE, 128, 0, stream>>>(dist, eidx, zat,
                                     (const float*)d_in[5], (const float*)d_in[6],
                                     (const float*)d_in[7], (const float*)d_in[8],
                                     (const float*)d_in[9], (const float*)d_in[10], xe, NE);
  rotate_fwd<<<NE, 256, 0, stream>>>(x, wig, eidx, NE, xmsrc, xmtgt);
  hipMemsetAsync(msg, 0, (size_t)NE * RED_STRIDE * 4, stream);

  for (int s = 0; s < 2; ++s) {
    const bf16* xm = s ? xmtgt : xmsrc;
    int B = 11 + s * 16;
    auto W = [&](int i) { return (const float*)d_in[i]; };
    // m = 0 : K = 7*128 = 896
    so2_gemm1<<<dim3(cdiv(NE, 64), 2), 256, 0, stream>>>(xm, NE, 1, 0, 0, 896, W(B + 2), T);
    so2_gate<<<NE, 256, 0, stream>>>(T, xe, W(B + 0), W(B + 1), 0, 1);
    so2_gemm2<<<dim3(cdiv(NE, 64), 7), 256, 0, stream>>>(T, NE, 896, W(B + 3), msg, 1, 0, 0, 0);
    // m = 1 : K = 6*128 = 768, rows = 2E (real/imag components interleaved)
    so2_gemm1<<<dim3(cdiv(2 * NE, 64), 2), 256, 0, stream>>>(xm, 2 * NE, 2, 896, 768, 768,
                                                            W(B + 6), T);
    so2_gate<<<NE, 256, 0, stream>>>(T, xe, W(B + 4), W(B + 5), 0, 2);
    so2_gemm2<<<dim3(cdiv(2 * NE, 64), 6), 256, 0, stream>>>(T, 2 * NE, 768, W(B + 7), msg, 2,
                                                             896, 768, 0);
    so2_gemm1<<<dim3(cdiv(2 * NE, 64), 2), 256, 0, stream>>>(xm, 2 * NE, 2, 896, 768, 768,
                                                            W(B + 8), T);
    so2_gate<<<NE, 256, 0, stream>>>(T, xe, W(B + 4), W(B + 5), 256, 2);
    so2_gemm2<<<dim3(cdiv(2 * NE, 64), 6), 256, 0, stream>>>(T, 2 * NE, 768, W(B + 9), msg, 2,
                                                             896, 768, 1);
    // m = 2 : K = 5*128 = 640
    so2_gemm1<<<dim3(cdiv(2 * NE, 64), 2), 256, 0, stream>>>(xm, 2 * NE, 2, 2432, 640, 640,
                                                            W(B + 12), T);
    so2_gate<<<NE, 256, 0, stream>>>(T, xe, W(B + 10), W(B + 11), 0, 2);
    so2_gemm2<<<dim3(cdiv(2 * NE, 64), 5), 256, 0, stream>>>(T, 2 * NE, 640, W(B + 13), msg, 2,
                                                             2432, 640, 0);
    so2_gemm1<<<dim3(cdiv(2 * NE, 64), 2), 256, 0, stream>>>(xm, 2 * NE, 2, 2432, 640, 640,
                                                            W(B + 14), T);
    so2_gate<<<NE, 256, 0, stream>>>(T, xe, W(B + 10), W(B + 11), 256, 2);
    so2_gemm2<<<dim3(cdiv(2 * NE, 64), 5), 256, 0, stream>>>(T, 2 * NE, 640, W(B + 15), msg, 2,
                                                             2432, 640, 1);
  }

  grid_mm<<<NE, 256, 0, stream>>>(msg, tgmm, fgmm, msg2);
  hipMemsetAsync(nmsg, 0, (size_t)NN * FULL_STRIDE * 4, stream);
  winv_scatter<<<NE, 256, 0, stream>>>(msg2, wig, eidx, NE, nmsg);
  node_sphere<<<NN, 256, 0, stream>>>(x, nmsg, tgll, fgll,
                                      (const float*)d_in[43], (const float*)d_in[44],
                                      (const float*)d_in[45], (const float*)d_in[46],
                                      (const float*)d_in[47], (const float*)d_in[48],
                                      (float*)d_out);
  (void)n_in; (void)out_size; (void)ws_size;
}
